// MovieNet_75892072120377
// MI455X (gfx1250) — matrix operand order repaired and checked by reading, not compiled, not run on hardware
//
#include <hip/hip_runtime.h>

typedef __attribute__((ext_vector_type(16))) __bf16 v16bf;
typedef __attribute__((ext_vector_type(8)))  __bf16 v8bf;
typedef __attribute__((ext_vector_type(8)))  float  v8f;

union V16 { v16bf v; v8bf h[2]; };

#define NB 32
#define NS 512
#define NH 256

__constant__ int d_pairA[10] = {0,0,0,0,1,1,1,2,2,3};
__constant__ int d_pairB[10] = {1,2,3,4,2,3,4,3,4,4};
__constant__ int d_uniJ[5]   = {0,1,2,3,3};
__constant__ int d_Klen[5]   = {320,256,192,288,224};

__device__ inline unsigned short f2bf(float f){
  unsigned int u = __float_as_uint(f);
  u += 0x7FFFu + ((u >> 16) & 1u);
  return (unsigned short)(u >> 16);
}
__device__ inline float sigm(float x){ return 1.0f/(1.0f+__expf(-x)); }
__device__ inline float lrelu(float x){ return x > 0.f ? x : 0.01f*x; }

// ---------------------------------------------------------------- fold
// folded[d][k] = sum_h W_r[k,h] * w_d[h]; bconst[d] = b_r . w_d
__global__ void fold_kernel(const float* W_f, const float* W_t, const float* W_c,
                            const float* W_s, const float* W_s2,
                            const float* b_f, const float* b_t, const float* b_c,
                            const float* b_s,
                            const float* Wa, const float* Wu,
                            float* folded, float* bconst)
{
  int d = blockIdx.x;
  int r; const float* vec;
  if (d < 20){ int i = d >> 1, slot = d & 1;
    r = slot ? d_pairB[i] : d_pairA[i];
    vec = Wa + i*2*NH + slot*NH;
  } else { r = d - 20; vec = Wu + d_uniJ[r]*NH; }
  const float* W  = (r==0)?W_f:(r==1)?W_t:(r==2)?W_c:(r==3)?W_s:W_s2;
  const float* bb = (r==0)?b_f:(r==1)?b_t:(r==2)?b_c:b_s;  // P and O share b_s
  int K = d_Klen[r];
  int k = threadIdx.x;
  if (k < K){
    float acc = 0.f;
    for (int h2 = 0; h2 < NH; ++h2) acc += W[k*NH + h2]*vec[h2];
    folded[d*320 + k] = acc;
  }
  if (k == 0){
    float acc = 0.f;
    for (int h2 = 0; h2 < NH; ++h2) acc += bb[h2]*vec[h2];
    bconst[d] = acc;
  }
}

// ---------------------------------------------------------------- scores
// jobs 0..9: pair attention logits; jobs 10..14: unimodal logits.
// scores layout: [15][B][S]
__global__ void scores_kernel(const float* Fr, const float* Te, const float* Ce,
                              const float* Pa, const float* Oc,
                              const float* folded, const float* bconst,
                              const float* ba, const float* bu, float* scores)
{
  int row = blockIdx.x;                 // b*S + s
  int b = row >> 9, s = row & 511;
  int tid = threadIdx.x, wave = tid >> 5, lane = tid & 31;
  const float* X[5] = { Fr + row*320, Te + row*256, Ce + row*192,
                        Pa + row*288, Oc + row*224 };
  for (int job = wave; job < 15; job += 8){
    float part = 0.f, cst;
    if (job < 10){
      int i = job, ra = d_pairA[i], rb = d_pairB[i];
      int Ka = d_Klen[ra], Kb = d_Klen[rb];
      const float* u = folded + (2*i)*320;
      const float* v = folded + (2*i+1)*320;
      const float* Xa = X[ra]; const float* Xb = X[rb];
      for (int k = lane; k < Ka + Kb; k += 32)
        part += (k < Ka) ? Xa[k]*u[k] : Xb[k-Ka]*v[k-Ka];
      cst = bconst[2*i] + bconst[2*i+1] + ba[i];
    } else {
      int r = job - 10, K = d_Klen[r];
      const float* u = folded + (20+r)*320;
      const float* Xr = X[r];
      for (int k = lane; k < K; k += 32) part += Xr[k]*u[k];
      cst = bconst[20+r] + bu[d_uniJ[r]];
    }
    for (int off = 16; off > 0; off >>= 1) part += __shfl_xor(part, off, 32);
    if (lane == 0) scores[(job*NB + b)*NS + s] = lrelu(part + cst);
  }
}

// ---------------------------------------------------------------- softmax over S, in place
__global__ void softmax_kernel(float* scores){
  int job = blockIdx.x >> 5, b = blockIdx.x & 31;
  float* rowp = scores + (job*NB + b)*NS;
  int tid = threadIdx.x;
  __shared__ float red[256];
  float v0 = rowp[tid], v1 = rowp[tid + 256];
  red[tid] = fmaxf(v0, v1); __syncthreads();
  for (int o = 128; o > 0; o >>= 1){
    if (tid < o) red[tid] = fmaxf(red[tid], red[tid+o]);
    __syncthreads();
  }
  float M = red[0]; __syncthreads();
  float e0 = __expf(v0 - M), e1 = __expf(v1 - M);
  red[tid] = e0 + e1; __syncthreads();
  for (int o = 128; o > 0; o >>= 1){
    if (tid < o) red[tid] += red[tid+o];
    __syncthreads();
  }
  float inv = 1.0f / red[0];
  rowp[tid] = e0*inv; rowp[tid + 256] = e1*inv;
}

// ---------------------------------------------------------------- weight prep (bf16, [N][K])
__global__ void prep_whT(const float* Wh, unsigned short* WhT){
  int id = blockIdx.x*blockDim.x + threadIdx.x;   // 1024*256
  int n = id >> 8, k = id & 255;
  WhT[n*256 + k] = f2bf(Wh[k*1024 + n]);
}
// Wi_dec^T, K padded 258->288: WiT[1024][288]
__global__ void prep_wiT_dec(const float* Wi, unsigned short* WiT){
  int id = blockIdx.x*blockDim.x + threadIdx.x;   // 1024*288
  if (id >= 1024*288) return;
  int n = id / 288, kk = id - n*288;
  WiT[n*288 + kk] = f2bf(kk < 258 ? Wi[kk*1024 + n] : 0.f);
}

// ---------------------------------------------------------------- encoder input precompute
// zx[b,s,col] = b_enc[col] + sum_k enc_in[b,s,k]*Wi_enc[k,col]; enc_in via reshape map
__global__ void enc_zx_kernel(const float* scores, const float* Wi, const float* bvec,
                              float* zx)
{
  int row = blockIdx.x;                 // b*S + s
  int b = row >> 9, s = row & 511;
  int tid = threadIdx.x;                // 256
  __shared__ float xv[8];
  if (tid < 5){
    int flat = s*5 + tid, q = flat >> 9, rr = flat & 511;
    xv[tid] = scores[((10+q)*NB + b)*NS + rr];
  }
  __syncthreads();
  float x0 = xv[0], x1 = xv[1], x2 = xv[2], x3 = xv[3], x4 = xv[4];
  #pragma unroll
  for (int i = 0; i < 4; ++i){
    int col = tid + i*256;
    zx[(long)row*1024 + col] = bvec[col]
      + x0*Wi[col] + x1*Wi[1024+col] + x2*Wi[2048+col]
      + x3*Wi[3072+col] + x4*Wi[4096+col];
  }
}

// ---------------------------------------------------------------- decoder input GEMM (hoisted off the critical path)
// zx[row, n] = b_dec[n] + sum_k decX[row,k]*Wi_dec[k,n]   (bf16 WMMA, K=288)
__global__ void __launch_bounds__(512)
decx_gemm(const unsigned short* decX, const unsigned short* WiT,
          const float* b_dec, float* zx)
{
  int tid = threadIdx.x;
  int wave = tid >> 5, lane = tid & 31, hi = lane >> 4, lo = lane & 15;
  int sid = blockIdx.x*16 + wave;       // 16384 strips: 1024 M-strips x 16 N-strips
  int m0 = (sid >> 4)*16, n0 = (sid & 15)*64;
  const unsigned short* arow = decX + (long)(m0 + lo)*288;
  v8f acc[4] = {};
  #pragma unroll
  for (int kt = 0; kt < 9; ++kt){
    int kb = kt*32;
    V16 A;
    A.h[0] = *(const v8bf*)&arow[kb + hi*8];
    A.h[1] = *(const v8bf*)&arow[kb + 16 + hi*8];
    #pragma unroll
    for (int t = 0; t < 4; ++t){
      int n = n0 + t*16 + lo;
      V16 Bv; Bv.h[0] = *(const v8bf*)&WiT[n*288 + kb + hi*8]; Bv.h[1] = *(const v8bf*)&WiT[n*288 + kb + 16 + hi*8];
      acc[t] = __builtin_amdgcn_wmma_f32_16x16x32_bf16(
          false, A.v, false, Bv.v, (short)0, acc[t], false, false);
    }
  }
  #pragma unroll
  for (int t = 0; t < 4; ++t){
    int n = n0 + t*16 + lo;
    float bd = b_dec[n];
    #pragma unroll
    for (int r = 0; r < 8; ++r)
      zx[(long)(m0 + r + 8*hi)*1024 + n] = acc[t][r] + bd;
  }
}

// ---------------------------------------------------------------- sequential LSTM core (recurrent part only, K=256)
// 2 blocks x 16 batch rows; 16 waves. Wave q owns N-tiles {q,16+q,32+q,48+q},
// so acc[g] holds gate g pre-activations for columns j=q*16+lo, rows m=r+8*hi.
// Gates computed directly from WMMA accumulators: no z round-trip, c in regs,
// h ping-pong buffered in LDS -> one barrier per step.
__global__ void __launch_bounds__(512)
lstm_seq(const float* zx, const unsigned short* WhT,
         const float* h0v, const float* c0v, float* hout)
{
  __shared__ __align__(32) unsigned short sh_h[2*16*256];  // ping-pong h (bf16)
  int tid = threadIdx.x;
  int wave = tid >> 5, lane = tid & 31, hi = lane >> 4, lo = lane & 15;
  int b0 = blockIdx.x * 16;
  int jcol = wave*16 + lo;                    // this lane's gate column j

  // init h buffer 0 and register-resident c
  for (int idx = tid; idx < 4096; idx += 512){
    int j = idx & 255;
    sh_h[idx] = h0v ? f2bf(h0v[j]) : (unsigned short)0;
  }
  float creg[8];
  #pragma unroll
  for (int r = 0; r < 8; ++r) creg[r] = c0v ? c0v[jcol] : 0.f;
  __syncthreads();

  // invariant B row pointers (4 gate tiles for this lane's column)
  const unsigned short* Bp[4];
  #pragma unroll
  for (int g = 0; g < 4; ++g) Bp[g] = WhT + (g*256 + jcol)*256;
  // invariant zx row offsets for the 8 rows this lane owns
  const float* zrow[8];
  #pragma unroll
  for (int r = 0; r < 8; ++r)
    zrow[r] = zx + (long)(b0 + r + 8*hi)*NS*1024 + jcol;

  for (int s = 0; s < NS; ++s){
    const unsigned short* hsrc = sh_h + (s & 1)*4096;
    unsigned short* hdst = sh_h + ((s & 1) ^ 1)*4096;

    // prefetch gate inputs for this step (independent of recurrence;
    // completes under the WMMA phase)
    float zxr[32];
    #pragma unroll
    for (int g = 0; g < 4; ++g)
      #pragma unroll
      for (int r = 0; r < 8; ++r)
        zxr[g*8+r] = zrow[r][(long)s*1024 + g*256];

    v8f acc[4] = {};
    #pragma unroll
    for (int kt = 0; kt < 8; ++kt){
      int kb = kt*32;
      V16 A;
      A.h[0] = *(const v8bf*)&hsrc[lo*256 + kb + hi*8];
      A.h[1] = *(const v8bf*)&hsrc[lo*256 + kb + 16 + hi*8];
      #pragma unroll
      for (int g = 0; g < 4; ++g){
        V16 Bv; Bv.h[0] = *(const v8bf*)&Bp[g][kb + hi*8]; Bv.h[1] = *(const v8bf*)&Bp[g][kb + 16 + hi*8];
        acc[g] = __builtin_amdgcn_wmma_f32_16x16x32_bf16(
            false, A.v, false, Bv.v, (short)0, acc[g], false, false);
      }
    }

    // LSTM cell update straight from accumulators
    #pragma unroll
    for (int r = 0; r < 8; ++r){
      int m = r + 8*hi;
      float zi = acc[0][r] + zxr[r];
      float zf = acc[1][r] + zxr[8+r];
      float zg = acc[2][r] + zxr[16+r];
      float zo = acc[3][r] + zxr[24+r];
      float cn = sigm(zf)*creg[r] + sigm(zi)*tanhf(zg);
      float hn = sigm(zo)*tanhf(cn);
      creg[r] = cn;
      hdst[m*256 + jcol] = f2bf(hn);
      hout[((b0+m)*NS + s)*NH + jcol] = hn;
    }
    __syncthreads();
  }
}

// ---------------------------------------------------------------- context + decoder input (bf16, K padded to 288)
__global__ void context_decx(const float* scores, const float* enc_out,
                             const float* t0, const float* t1, unsigned short* decX)
{
  int row = blockIdx.x;                 // b*S + s
  int b = row >> 9, s = row & 511;
  int tid = threadIdx.x;                // 256 = H
  __shared__ float aw[10];
  if (tid < 10){
    int flat = s*10 + tid, q = flat >> 9, rr = flat & 511;
    aw[tid] = scores[(q*NB + b)*NS + rr];
  }
  __syncthreads();
  float acc = 0.f;
  #pragma unroll
  for (int i = 0; i < 10; ++i)
    if (s - i >= 0) acc += aw[i]*enc_out[(row-i)*NH + tid];
  decX[row*288 + 2 + tid] = f2bf(acc);
  if (tid == 0) decX[row*288]     = f2bf(s > 0 ? t0[row-1] : 0.f);
  if (tid == 1) decX[row*288 + 1] = f2bf(s > 0 ? t1[row-1] : 0.f);
  if (tid >= 2 && tid < 32) decX[row*288 + 256 + tid] = 0;   // cols 258..287 pad
}

// ---------------------------------------------------------------- output head
__global__ void out_head(const float* dec_out, const float* Wo1, const float* bo1,
                         const float* Wo2, const float* bo2, float* out)
{
  int tid = threadIdx.x;
  int row = blockIdx.x*8 + (tid >> 5);
  int lane = tid & 31;
  float s0 = 0.f, s1 = 0.f;
  for (int h = lane; h < NH; h += 32){
    float d = dec_out[row*NH + h];
    s0 += d*Wo1[2*h]; s1 += d*Wo1[2*h+1];
  }
  for (int off = 16; off > 0; off >>= 1){
    s0 += __shfl_xor(s0, off, 32);
    s1 += __shfl_xor(s1, off, 32);
  }
  float u0 = lrelu(s0 + bo1[0]);
  float u1 = lrelu(s1 + bo1[1]);
  if (lane < 9) out[row*9 + lane] = u0*Wo2[lane] + u1*Wo2[9+lane] + bo2[lane];
}

// ---------------------------------------------------------------- launch
extern "C" void kernel_launch(void* const* d_in, const int* in_sizes, int n_in,
                              void* d_out, int out_size, void* d_ws, size_t ws_size,
                              hipStream_t stream)
{
  (void)in_sizes; (void)n_in; (void)out_size; (void)ws_size;
  const float* Fr     = (const float*)d_in[1];
  const float* Te     = (const float*)d_in[2];
  const float* Ce     = (const float*)d_in[3];
  const float* Pa     = (const float*)d_in[4];
  const float* Oc     = (const float*)d_in[5];
  const float* t0     = (const float*)d_in[6];
  const float* t1     = (const float*)d_in[7];
  const float* W_f    = (const float*)d_in[8];
  const float* b_f    = (const float*)d_in[9];
  const float* W_t    = (const float*)d_in[10];
  const float* b_t    = (const float*)d_in[11];
  const float* W_c    = (const float*)d_in[12];
  const float* b_c    = (const float*)d_in[13];
  const float* W_s    = (const float*)d_in[14];
  const float* b_s    = (const float*)d_in[15];
  const float* W_s2   = (const float*)d_in[16];
  const float* Wa     = (const float*)d_in[17];
  const float* ba     = (const float*)d_in[18];
  const float* Wu     = (const float*)d_in[19];
  const float* bu     = (const float*)d_in[20];
  const float* Wi_enc = (const float*)d_in[21];
  const float* Wh_enc = (const float*)d_in[22];
  const float* b_enc  = (const float*)d_in[23];
  const float* Wi_dec = (const float*)d_in[24];
  const float* Wh_dec = (const float*)d_in[25];
  const float* b_dec  = (const float*)d_in[26];
  const float* dec_h0 = (const float*)d_in[27];
  const float* dec_c0 = (const float*)d_in[28];
  const float* Wo1    = (const float*)d_in[29];
  const float* bo1    = (const float*)d_in[30];
  const float* Wo2    = (const float*)d_in[31];
  const float* bo2    = (const float*)d_in[32];

  char* ws = (char*)d_ws;
  size_t off = 0;
  auto alloc = [&](size_t bytes)->char* {
    char* p = ws + off;
    off += (bytes + 255) & ~(size_t)255;
    return p;
  };
  float* folded          = (float*)alloc(25*320*sizeof(float));
  float* bconst          = (float*)alloc(25*sizeof(float));
  float* scores          = (float*)alloc((size_t)15*NB*NS*sizeof(float));
  unsigned short* WhTe   = (unsigned short*)alloc((size_t)1024*256*2);
  unsigned short* WhTd   = (unsigned short*)alloc((size_t)1024*256*2);
  unsigned short* WiTd   = (unsigned short*)alloc((size_t)1024*288*2);
  float* enc_out         = (float*)alloc((size_t)NB*NS*NH*sizeof(float));
  unsigned short* decX   = (unsigned short*)alloc((size_t)NB*NS*288*2);
  float* zxe             = (float*)alloc((size_t)NB*NS*1024*sizeof(float));
  float* zxd             = (float*)alloc((size_t)NB*NS*1024*sizeof(float));
  float* dec_outb        = (float*)alloc((size_t)NB*NS*NH*sizeof(float));

  fold_kernel<<<25, 320, 0, stream>>>(W_f, W_t, W_c, W_s, W_s2,
                                      b_f, b_t, b_c, b_s, Wa, Wu, folded, bconst);
  prep_whT<<<1024, 256, 0, stream>>>(Wh_enc, WhTe);
  prep_whT<<<1024, 256, 0, stream>>>(Wh_dec, WhTd);
  prep_wiT_dec<<<(1024*288 + 255)/256, 256, 0, stream>>>(Wi_dec, WiTd);
  scores_kernel<<<NB*NS, 256, 0, stream>>>(Fr, Te, Ce, Pa, Oc,
                                           folded, bconst, ba, bu, scores);
  softmax_kernel<<<15*NB, 256, 0, stream>>>(scores);
  enc_zx_kernel<<<NB*NS, 256, 0, stream>>>(scores, Wi_enc, b_enc, zxe);
  lstm_seq<<<2, 512, 0, stream>>>(zxe, WhTe, nullptr, nullptr, enc_out);
  context_decx<<<NB*NS, 256, 0, stream>>>(scores, enc_out, t0, t1, decX);
  decx_gemm<<<1024, 512, 0, stream>>>(decX, WiTd, b_dec, zxd);
  lstm_seq<<<2, 512, 0, stream>>>(zxd, WhTd, dec_h0, dec_c0, dec_outb);
  out_head<<<NB*NS/8, 256, 0, stream>>>(dec_outb, Wo1, bo1, Wo2, bo2, (float*)d_out);
}
